// MultiHeadAttention_54022098649169
// MI455X (gfx1250) — matrix operand order repaired and checked by reading, not compiled, not run on hardware
//
#include <hip/hip_runtime.h>
#include <hip/hip_bf16.h>

// MultiHeadAttention forward for MI455X (gfx1250, wave32, WMMA + TDM).
//
// Sizes: S=1024, B=4, D=1024, H=16, DK=DV=64, D_OUT=1024.
// All matrix math on v_wmma_f32_16x16x32_f16 (f16 in, f32 accum).
// d_out = [ out : 4,194,304 f32 | attn : 67,108,864 f32 ].
//
// Roofline: mandatory HBM writes ~285 MB (attn dominates) -> >= 12.3 us at
// 23.3 TB/s; total math ~52 GFLOP on the f16 WMMA pipe -> same order, so:
// f16 WMMA, register-blocked tiles, f16-compact intermediates (L2-resident),
// and the attn strip streamed LDS->HBM by the Tensor Data Mover so the
// dominant store overlaps the attn@V WMMA phase.

#define S_LEN 1024
#define BATCH 4
#define DMODEL 1024
#define NHEAD 16
#define DK 64
#define ROWS (S_LEN * BATCH)          // 4096 GEMM rows (r = s*B + b)

typedef __attribute__((ext_vector_type(16))) _Float16 v16h;
typedef __attribute__((ext_vector_type(8)))  float    v8f;
typedef __attribute__((ext_vector_type(4)))  unsigned int u32x4;
typedef __attribute__((ext_vector_type(8)))  unsigned int u32x8;

// ---------------------------------------------------------------------------
// WMMA helpers.  Fragment layouts per CDNA5 ISA §7.12.2 (wave32):
//   A (16x32 f16): lane 0-15 row M=lane, lane 16-31 row M=lane-16 (hi=lane/16
//     selects K chunks): per-lane K = {k0+hi*8 .. +7} U {k0+16+hi*8 .. +7}
//   B (32x16 f16): col N = lane&15, per-lane K = {k0+hi*8 .. +7} U {k0+16+hi*8 .. +7} (as A)
//   C/D (16x16 f32, 8 VGPRs): elem r -> M = r + 8*(lane/16), N = lane&15
// ---------------------------------------------------------------------------
__device__ __forceinline__ v8f wmma16(v16h a, v16h b, v8f c) {
  return __builtin_amdgcn_wmma_f32_16x16x32_f16(
      /*neg_a=*/false, a, /*neg_b=*/false, b,
      /*c_mod=*/(short)0, c, /*reuse_a=*/false, /*reuse_b=*/false);
}

__device__ __forceinline__ v16h load_a_frag(const _Float16* A, int lda,
                                            int row, int k0, int hi) {
  const _Float16* p = A + row * lda + k0 + hi * 8;
  v16h a;
#pragma unroll
  for (int e = 0; e < 8; ++e) {
    a[e]     = p[e];        // K = k0 + hi*8 + e
    a[e + 8] = p[16 + e];   // K = k0 + 16 + hi*8 + e
  }
  return a;
}

// Bt is the N x K (row-major) view of the K x N matrix B.
__device__ __forceinline__ v16h load_b_frag(const _Float16* Bt, int ldb,
                                            int col, int k0, int hi) {
  const _Float16* p = Bt + col * ldb + k0 + hi * 8;
  v16h b;
#pragma unroll
  for (int e = 0; e < 8; ++e) {
    b[e]     = p[e];        // K = k0 + hi*8 + e
    b[e + 8] = p[16 + e];   // K = k0 + 16 + hi*8 + e
  }
  return b;
}

// ---------------------------------------------------------------------------
// Stage 1a: fp32 -> f16 elementwise (activations)
// ---------------------------------------------------------------------------
__global__ void cvt_f32_to_f16(const float* __restrict__ x,
                               _Float16* __restrict__ y, int n) {
  int i = blockIdx.x * blockDim.x + threadIdx.x;
  if (i < n) y[i] = (_Float16)x[i];
}

// ---------------------------------------------------------------------------
// Stage 1b: W[K][N] fp32 -> Wt[N][K] f16, LDS-tiled transpose (32x32)
// ---------------------------------------------------------------------------
__global__ void transpose_to_f16(const float* __restrict__ W,
                                 _Float16* __restrict__ Wt, int K, int N) {
  __shared__ float tile[32][33];
  int kk = blockIdx.y * 32 + threadIdx.y;
  int nn = blockIdx.x * 32 + threadIdx.x;
  tile[threadIdx.y][threadIdx.x] = W[kk * N + nn];
  __syncthreads();
  int on = blockIdx.x * 32 + threadIdx.y;
  int ok = blockIdx.y * 32 + threadIdx.x;
  Wt[on * K + ok] = (_Float16)tile[threadIdx.x][threadIdx.y];
}

// ---------------------------------------------------------------------------
// Stage 2 / 4: register-blocked GEMM, one 32x64 output tile per wave
// (2 A-frags + 4 B-frags -> 8 WMMAs per 32-wide k-step, 64 accum VGPRs).
//   C = A(MxK f16 row-major) * Bt^T + bias
// MODE 0: q/k projection -> f16 [H][B][S][DK]
// MODE 1: v   projection -> f16 [H][B][DV][S]   (transposed for attn@V)
// MODE 2: out projection -> f32 row-major [M][N] (+bias)
// ---------------------------------------------------------------------------
template <int MODE>
__global__ void gemm32x64(const _Float16* __restrict__ A,
                          const _Float16* __restrict__ Bt,
                          const float* __restrict__ bias,
                          void* __restrict__ Cout, int M, int N, int K) {
  const int lane = threadIdx.x & 31;
  const int col  = lane & 15;
  const int hi   = lane >> 4;
  const int wid  = blockIdx.x * (blockDim.x >> 5) + (threadIdx.x >> 5);
  const int nt   = N >> 6;              // 64-wide column tiles
  const int tm   = wid / nt;
  const int tn   = wid % nt;
  if (tm >= (M >> 5)) return;
  const int m0 = tm << 5, n0 = tn << 6;

  v8f acc[2][4];
#pragma unroll
  for (int i = 0; i < 2; ++i)
#pragma unroll
    for (int j = 0; j < 4; ++j) acc[i][j] = (v8f){};

  for (int k0 = 0; k0 < K; k0 += 32) {
    if (k0 + 32 < K) {   // gfx1250 global_prefetch_b8 for the next k-step
      __builtin_prefetch(A + (size_t)(m0 + col) * K + k0 + 32, 0, 0);
      __builtin_prefetch(Bt + (size_t)(n0 + col) * K + k0 + 32, 0, 0);
    }
    v16h a0 = load_a_frag(A, K, m0 + col, k0, hi);
    v16h a1 = load_a_frag(A, K, m0 + 16 + col, k0, hi);
#pragma unroll
    for (int j = 0; j < 4; ++j) {
      v16h b = load_b_frag(Bt, K, n0 + j * 16 + col, k0, hi);
      acc[0][j] = wmma16(a0, b, acc[0][j]);
      acc[1][j] = wmma16(a1, b, acc[1][j]);
    }
  }

#pragma unroll
  for (int i = 0; i < 2; ++i) {
#pragma unroll
    for (int j = 0; j < 4; ++j) {
      const int n = n0 + j * 16 + col;
      const float bn = bias[n];
#pragma unroll
      for (int r = 0; r < 8; ++r) {
        const int m = m0 + i * 16 + r + (hi << 3);
        const float val = acc[i][j][r] + bn;
        if (MODE == 2) {
          ((float*)Cout)[(size_t)m * N + n] = val;
        } else {
          const int s = m >> 2, bb = m & 3;   // row = s*B + b
          const int h = n >> 6, d = n & 63;   // col = h*64 + d
          if (MODE == 0)   // [H][B][S][DK]
            ((_Float16*)Cout)[(((size_t)(h * BATCH + bb) * S_LEN) + s) * DK + d] =
                (_Float16)val;
          else             // [H][B][DV][S]
            ((_Float16*)Cout)[(((size_t)(h * BATCH + bb) * DK) + d) * S_LEN + s] =
                (_Float16)val;
        }
      }
    }
  }
}

// ---------------------------------------------------------------------------
// Stage 3: attention.  One 4-wave block per (h, b, 16-row s-tile), sharing a
// 64 KB fp32 logits strip in LDS:
//   phase 1: wave w computes logits columns [w*256, w*256+256)      (32 WMMA)
//   phase 2: wave w softmaxes rows [w*4, w*4+4) in place (LDS)
//   TDM    : wave 0 issues tensor_store_from_lds of the contiguous 64 KB
//            attn tile (rows m0..m0+15 are adjacent in d_out) -> overlaps
//            with phase 3; s_wait_tensorcnt 0 before kernel exit
//   phase 3: wave w computes ctx column tile dv = [w*16, w*16+16)   (32 WMMA)
// ---------------------------------------------------------------------------
__global__ void attention_kernel(const _Float16* __restrict__ q,
                                 const _Float16* __restrict__ k,
                                 const _Float16* __restrict__ vT,
                                 float* __restrict__ attn_out,
                                 _Float16* __restrict__ ctx) {
  __shared__ float slog[16 * S_LEN];   // 64 KB

  const int lane = threadIdx.x & 31;
  const int wave = threadIdx.x >> 5;   // 0..3
  const int col  = lane & 15;
  const int hi   = lane >> 4;
  const int wid  = blockIdx.x;
  const int hb   = wid >> 6;           // h*B + b
  const int m0   = (wid & 63) << 4;    // s-tile origin
  const int h    = hb >> 2;
  const int bb   = hb & 3;

  const _Float16* qb  = q  + (size_t)hb * S_LEN * DK;
  const _Float16* kb  = k  + (size_t)hb * S_LEN * DK;
  const _Float16* vTb = vT + (size_t)hb * DK * S_LEN;

  // ---- phase 1: logits slice (16 n-tiles x 2 WMMAs per wave) --------------
  v16h aq0 = load_a_frag(qb, DK, m0 + col, 0, hi);
  v16h aq1 = load_a_frag(qb, DK, m0 + col, 32, hi);
  for (int t0 = wave * 256; t0 < wave * 256 + 256; t0 += 16) {
    v8f c = {};
    c = wmma16(aq0, load_b_frag(kb, DK, t0 + col, 0, hi), c);
    c = wmma16(aq1, load_b_frag(kb, DK, t0 + col, 32, hi), c);
#pragma unroll
    for (int r = 0; r < 8; ++r)
      slog[(r + (hi << 3)) * S_LEN + t0 + col] = c[r] * 0.125f;  // 1/sqrt(64)
  }
  __syncthreads();

  // ---- phase 2: softmax (4 rows per wave), normalized in place ------------
  for (int m = wave * 4; m < wave * 4 + 4; ++m) {
    float* row = slog + m * S_LEN;
    float lmax = -3.0e38f;
    for (int n = lane; n < S_LEN; n += 32) lmax = fmaxf(lmax, row[n]);
#pragma unroll
    for (int off = 16; off > 0; off >>= 1)
      lmax = fmaxf(lmax, __shfl_xor(lmax, off, 32));
    float lsum = 0.f;
    for (int n = lane; n < S_LEN; n += 32) {
      float e = __expf(row[n] - lmax);
      row[n] = e;
      lsum += e;
    }
#pragma unroll
    for (int off = 16; off > 0; off >>= 1)
      lsum += __shfl_xor(lsum, off, 32);
    const float inv = 1.0f / lsum;
    for (int n = lane; n < S_LEN; n += 32) row[n] *= inv;
  }
  __syncthreads();

  // ---- TDM: DMA the contiguous 16x1024 f32 attn tile LDS -> HBM -----------
  // D# per CDNA5 ISA: 1-D tensor, data_size=4B,
  // tensor_dim0 = tile_dim0 = 16384 elements (64 KB).  VADDR2/3 = NULL.
  if (wave == 0) {
    unsigned long long ga =
        (unsigned long long)(attn_out + ((size_t)hb * S_LEN + m0) * S_LEN);
    unsigned lds_off = (unsigned)(unsigned long long)(&slog[0]);  // flat low32
    u32x4 g0;
    g0[0] = 1u;                                  // count=1, user descriptor
    g0[1] = lds_off;                             // lds_addr (bytes)
    g0[2] = (unsigned)(ga & 0xFFFFFFFFu);        // global_addr[31:0]
    g0[3] = (unsigned)((ga >> 32) & 0x01FFFFFFu) // global_addr[56:32]
            | (2u << 30);                        // type = 2 ("image")
    u32x8 g1;
    g1[0] = 2u << 16;                            // data_size = 4 bytes
    g1[1] = (16384u & 0xFFFFu) << 16;            // tensor_dim0[15:0]
    g1[2] = (16384u >> 16) | (1u << 16);         // tensor_dim0[31:16], dim1=1
    g1[3] = 16384u << 16;                        // tile_dim0 = 16384
    g1[4] = 1u;                                  // tile_dim1 = 1
    g1[5] = 16384u;                              // tensor_dim0_stride lo32
    g1[6] = 0u;
    g1[7] = 0u;
    asm volatile("tensor_store_from_lds %0, %1" :: "s"(g0), "s"(g1)
                 : "memory");
  }

  // ---- phase 3 (overlaps TDM store; slog is read-only from here) ----------
  v8f cc = {};
  for (int k0 = 0; k0 < S_LEN; k0 += 32) {
    v16h a;
    const float* arow = slog + col * S_LEN + k0 + (hi << 3);
#pragma unroll
    for (int e = 0; e < 8; ++e) {
      a[e]     = (_Float16)arow[e];
      a[e + 8] = (_Float16)arow[16 + e];
    }
    cc = wmma16(a, load_b_frag(vTb, S_LEN, wave * 16 + col, k0, hi), cc);
  }
#pragma unroll
  for (int r = 0; r < 8; ++r) {
    const int s = m0 + r + (hi << 3);
    ctx[((size_t)s * BATCH + bb) * (NHEAD * DK) + h * DK + wave * 16 + col] =
        (_Float16)cc[r];
  }

  // Drain the tensor DMA before the workgroup (and its LDS) is retired.
  __builtin_amdgcn_s_wait_tensorcnt(0);
}

// ---------------------------------------------------------------------------
extern "C" void kernel_launch(void* const* d_in, const int* in_sizes, int n_in,
                              void* d_out, int out_size, void* d_ws,
                              size_t ws_size, hipStream_t stream) {
  const float* query = (const float*)d_in[0];
  const float* key   = (const float*)d_in[1];
  const float* value = (const float*)d_in[2];
  const float* Wq = (const float*)d_in[3];
  const float* bq = (const float*)d_in[4];
  const float* Wk = (const float*)d_in[5];
  const float* bk = (const float*)d_in[6];
  const float* Wv = (const float*)d_in[7];
  const float* bv = (const float*)d_in[8];
  const float* Wo = (const float*)d_in[9];
  const float* bo = (const float*)d_in[10];

  float* out_p  = (float*)d_out;                         // 4,194,304 f32
  float* attn_p = (float*)d_out + (size_t)ROWS * DMODEL; // 67,108,864 f32

  // workspace layout (f16 elements), 64 MB total
  _Float16* w = (_Float16*)d_ws;
  const size_t ACT = (size_t)ROWS * DMODEL;   // 4,194,304
  const size_t WT  = (size_t)DMODEL * DMODEL; // 1,048,576
  _Float16* qx  = w;              // activations f16
  _Float16* kx  = qx + ACT;
  _Float16* vx  = kx + ACT;
  _Float16* Wqt = vx + ACT;       // weights, N x K f16
  _Float16* Wkt = Wqt + WT;
  _Float16* Wvt = Wkt + WT;
  _Float16* Wot = Wvt + WT;
  _Float16* qp  = Wot + WT;       // q  [H][B][S][DK]
  _Float16* kp  = qp + ACT;       // k  [H][B][S][DK]
  _Float16* vtp = kp + ACT;       // v  [H][B][DV][S]
  _Float16* ctx = vtp + ACT;      // ctx [S*B][H*DV]

  // Stage 1: precision conversion + weight transposes
  {
    int n = (int)ACT;
    cvt_f32_to_f16<<<n / 256, 256, 0, stream>>>(query, qx, n);
    cvt_f32_to_f16<<<n / 256, 256, 0, stream>>>(key,   kx, n);
    cvt_f32_to_f16<<<n / 256, 256, 0, stream>>>(value, vx, n);
    dim3 tg(DMODEL / 32, DMODEL / 32), tb(32, 32);
    transpose_to_f16<<<tg, tb, 0, stream>>>(Wq, Wqt, DMODEL, DMODEL);
    transpose_to_f16<<<tg, tb, 0, stream>>>(Wk, Wkt, DMODEL, DMODEL);
    transpose_to_f16<<<tg, tb, 0, stream>>>(Wv, Wvt, DMODEL, DMODEL);
    transpose_to_f16<<<tg, tb, 0, stream>>>(Wo, Wot, DMODEL, DMODEL);
  }

  // Stage 2: QKV projections (2048 32x64 tiles, 4 waves / 128-thread block)
  {
    const int tiles  = (ROWS / 32) * (DMODEL / 64);   // 2048
    const int blocks = tiles / 4;                     // 512
    gemm32x64<0><<<blocks, 128, 0, stream>>>(qx, Wqt, bq, qp,  ROWS, DMODEL, DMODEL);
    gemm32x64<0><<<blocks, 128, 0, stream>>>(kx, Wkt, bk, kp,  ROWS, DMODEL, DMODEL);
    gemm32x64<1><<<blocks, 128, 0, stream>>>(vx, Wvt, bv, vtp, ROWS, DMODEL, DMODEL);
  }

  // Stage 3: attention (4-wave blocks; one (h,b,s-tile) each; 64 KB LDS)
  attention_kernel<<<NHEAD * BATCH * (S_LEN / 16), 128, 0, stream>>>(
      qp, kp, vtp, attn_p, ctx);

  // Stage 4: output projection, fp32 + bias -> d_out
  {
    const int tiles = (ROWS / 32) * (DMODEL / 64);
    gemm32x64<2><<<tiles / 4, 128, 0, stream>>>(ctx, Wot, bo, out_p,
                                                ROWS, DMODEL, DMODEL);
  }
}
